// SingleAttention_14139032338689
// MI455X (gfx1250) — matrix operand order repaired and checked by reading, not compiled, not run on hardware
//
#include <hip/hip_runtime.h>
#include <hip/hip_bf16.h>
#include <math.h>

// ---- problem constants (match reference) ----
#define DIMC 1024
#define NH   16
#define HD   64
#define BB   2
#define TT   2048
#define EPSF 1e-6f

typedef __attribute__((ext_vector_type(16))) __bf16 bf16x16;
typedef __attribute__((ext_vector_type(8)))  __bf16 bf16x8;
typedef __attribute__((ext_vector_type(8)))  float  f32x8;

__device__ __forceinline__ __bf16 f2bf(float f) {
  unsigned u = __builtin_bit_cast(unsigned, f);
  u += 0x7FFFu + ((u >> 16) & 1u);               // round-to-nearest-even
  unsigned short h = (unsigned short)(u >> 16);
  return __builtin_bit_cast(__bf16, h);
}

// ---------------------------------------------------------------------------
// f32 -> bf16 conversion
// ---------------------------------------------------------------------------
__global__ void attn_cvt_bf16(const float* __restrict__ in,
                              __bf16* __restrict__ out, int n) {
  int i = blockIdx.x * blockDim.x + threadIdx.x;
  if (i < n) out[i] = f2bf(in[i]);
}

// ---------------------------------------------------------------------------
// C[M,N] = A[M,K](bf16,row-major) @ W[N,K]^T (bf16,row-major), f32 out.
// One wave per 32x64 output tile; 8 wmma per K=32 step (2 A-frags x 4 B-frags).
// All loads of a K-step are hoisted ahead of the WMMA block so the scheduler
// can overlap them (partial s_wait_loadcnt) instead of draining per-fragment.
// ---------------------------------------------------------------------------
__global__ void attn_gemm_bf16(const __bf16* __restrict__ A,
                               const __bf16* __restrict__ W,
                               float* __restrict__ C,
                               int M, int N, int K) {
  const int lane = threadIdx.x;          // 0..31 (wave32)
  const int lo   = lane & 15;
  const int hi   = lane >> 4;
  const int m0   = blockIdx.y * 32;
  const int n0   = blockIdx.x * 64;
  const int ah   = hi * 16;              // A K-half offset (lanes 16-31 -> +16)
  const int bhh  = hi * 16;              // B K-half offset (lanes 16-31 -> +16)

  f32x8 acc[2][4] = {};
  const __bf16* Arow0 = A + (size_t)(m0 + lo) * K;        // rows m0..m0+15
  const __bf16* Arow1 = A + (size_t)(m0 + 16 + lo) * K;   // rows m0+16..m0+31
  const __bf16* Wr0   = W + (size_t)(n0 + lo) * K + bhh;  // col n0+lo
  const size_t  wstep = (size_t)16 * K;                   // +16 columns

  for (int k0 = 0; k0 < K; k0 += 32) {
    // ---- issue ALL loads for this K-step first ----
    bf16x8 l0 = *(const bf16x8*)(Arow0 + k0 + ah);
    bf16x8 h0 = *(const bf16x8*)(Arow0 + k0 + 8 + ah);
    bf16x8 l1 = *(const bf16x8*)(Arow1 + k0 + ah);
    bf16x8 h1 = *(const bf16x8*)(Arow1 + k0 + 8 + ah);
    bf16x16 bfr[4];
#pragma unroll
    for (int t = 0; t < 4; ++t)
      bfr[t] = *(const bf16x16*)(Wr0 + (size_t)t * wstep + k0);

    // ---- pack A fragments (register moves only) ----
    bf16x16 af0, af1;
#pragma unroll
    for (int i = 0; i < 8; ++i) {
      af0[i] = l0[i]; af0[8 + i] = h0[i];
      af1[i] = l1[i]; af1[8 + i] = h1[i];
    }

    // ---- 8 WMMA ----
#pragma unroll
    for (int t = 0; t < 4; ++t) {
      acc[0][t] = __builtin_amdgcn_wmma_f32_16x16x32_bf16(
          false, af0, false, bfr[t], (short)0, acc[0][t], false, false);
      acc[1][t] = __builtin_amdgcn_wmma_f32_16x16x32_bf16(
          false, af1, false, bfr[t], (short)0, acc[1][t], false, false);
    }
  }

  const int off8 = hi * 8; // C layout: lanes 16-31 hold rows M=8..15
#pragma unroll
  for (int half = 0; half < 2; ++half)
#pragma unroll
    for (int t = 0; t < 4; ++t) {
      const int n = n0 + t * 16 + lo;
#pragma unroll
      for (int r = 0; r < 8; ++r)
        C[(size_t)(m0 + half * 16 + r + off8) * N + n] = acc[half][t][r];
    }
}

// ---------------------------------------------------------------------------
// RMSNorm (q,k) + RoPE, write bf16 q/k [bh,T,hd] and v transposed [bh,hd,T].
// One wave per (b,h,t); lane d owns the rotation pair (d, d+32).
// ---------------------------------------------------------------------------
__global__ void attn_qkv_post(const float* __restrict__ qkv,
                              const float* __restrict__ qw,
                              const float* __restrict__ kw,
                              __bf16* __restrict__ qb,
                              __bf16* __restrict__ kb,
                              __bf16* __restrict__ vT) {
  const int idx = blockIdx.x;       // (b*NH + h)*TT + t
  const int bh  = idx / TT;
  const int t   = idx % TT;
  const int h   = bh % NH;
  const int b   = bh / NH;
  const int d   = threadIdx.x;      // 0..31

  const float* row = qkv + (size_t)(b * TT + t) * (3 * DIMC) + h * HD;
  float q0 = row[d],            q1 = row[d + 32];
  float k0 = row[DIMC + d],     k1 = row[DIMC + d + 32];
  float v0 = row[2 * DIMC + d], v1 = row[2 * DIMC + d + 32];

  float sq = q0 * q0 + q1 * q1;
  float sk = k0 * k0 + k1 * k1;
#pragma unroll
  for (int m = 1; m < 32; m <<= 1) {
    sq += __shfl_xor(sq, m, 32);
    sk += __shfl_xor(sk, m, 32);
  }
  const float rq = rsqrtf(sq * (1.0f / HD) + EPSF);
  const float rk = rsqrtf(sk * (1.0f / HD) + EPSF);
  q0 *= rq * qw[d]; q1 *= rq * qw[d + 32];
  k0 *= rk * kw[d]; k1 *= rk * kw[d + 32];

  // RoPE: freq_d = theta^(-d/32) = 2^(-5d/32), theta = 32
  const float ang = (float)t * exp2f(-5.0f * (float)d * (1.0f / 32.0f));
  float sn, cs;
  __sincosf(ang, &sn, &cs);
  const float qo0 = q0 * cs - q1 * sn, qo1 = q1 * cs + q0 * sn;
  const float ko0 = k0 * cs - k1 * sn, ko1 = k1 * cs + k0 * sn;

  const size_t qoff = ((size_t)bh * TT + t) * HD;
  qb[qoff + d]      = f2bf(qo0); qb[qoff + d + 32] = f2bf(qo1);
  kb[qoff + d]      = f2bf(ko0); kb[qoff + d + 32] = f2bf(ko1);
  vT[((size_t)bh * HD + d) * TT + t]      = f2bf(v0);
  vT[((size_t)bh * HD + d + 32) * TT + t] = f2bf(v1);
}

// ---------------------------------------------------------------------------
// Flash attention, causal. One wave per (b,h, 16-query tile).
// Streams 32-key chunks: 4 wmma QK^T + online softmax + 4 wmma PV.
// Output written bf16 to [b,t, h*HD+d] for the projection GEMM.
// ---------------------------------------------------------------------------
__global__ void attn_flash(const __bf16* __restrict__ Q,
                           const __bf16* __restrict__ Kt,
                           const __bf16* __restrict__ Vt,
                           __bf16* __restrict__ O) {
  const int bh   = blockIdx.x;      // 0..B*NH-1
  const int qt   = blockIdx.y;      // 0..TT/16-1
  const int lane = threadIdx.x;
  const int lo   = lane & 15;
  const int hi   = lane >> 4;
  const int qlo  = qt * 16;
  const int ah   = hi * 16;
  const int bh2  = hi * 16;
  const int off8 = hi * 8;
  const float scale = 0.125f;       // 1/sqrt(64)

  __shared__ __align__(16) __bf16 P[16 * 32];

  // Q A-fragments (16x64 -> two 16x32 frags), loaded once
  const __bf16* Qr = Q + ((size_t)bh * TT + qlo + lo) * HD;
  bf16x16 qa0, qa1;
  {
    bf16x8 l0 = *(const bf16x8*)(Qr + ah);
    bf16x8 h0 = *(const bf16x8*)(Qr + 8 + ah);
    bf16x8 l1 = *(const bf16x8*)(Qr + 32 + ah);
    bf16x8 h1 = *(const bf16x8*)(Qr + 40 + ah);
#pragma unroll
    for (int i = 0; i < 8; ++i) {
      qa0[i] = l0[i]; qa0[8 + i] = h0[i];
      qa1[i] = l1[i]; qa1[8 + i] = h1[i];
    }
  }

  float mstate[8], lstate[8];
#pragma unroll
  for (int r = 0; r < 8; ++r) { mstate[r] = -__builtin_inff(); lstate[r] = 0.f; }
  f32x8 o[4] = {};

  for (int j0 = 0; j0 <= qlo; j0 += 32) {
    // ---- issue all K-side loads for this chunk first ----
    const __bf16* Kr0 = Kt + ((size_t)bh * TT + j0 + lo) * HD;
    const __bf16* Kr1 = Kt + ((size_t)bh * TT + j0 + 16 + lo) * HD;
    bf16x16 kb00 = *(const bf16x16*)(Kr0 + bh2);
    bf16x16 kb01 = *(const bf16x16*)(Kr0 + 32 + bh2);
    bf16x16 kb10 = *(const bf16x16*)(Kr1 + bh2);
    bf16x16 kb11 = *(const bf16x16*)(Kr1 + 32 + bh2);

    // ---- S = Q K^T for 2 key sub-tiles of 16 ----
    f32x8 s[2];
    {
      f32x8 z0 = {}, z1 = {};
      z0 = __builtin_amdgcn_wmma_f32_16x16x32_bf16(false, qa0, false, kb00,
                                                   (short)0, z0, false, false);
      z1 = __builtin_amdgcn_wmma_f32_16x16x32_bf16(false, qa0, false, kb10,
                                                   (short)0, z1, false, false);
      z0 = __builtin_amdgcn_wmma_f32_16x16x32_bf16(false, qa1, false, kb01,
                                                   (short)0, z0, false, false);
      z1 = __builtin_amdgcn_wmma_f32_16x16x32_bf16(false, qa1, false, kb11,
                                                   (short)0, z1, false, false);
      s[0] = z0; s[1] = z1;
    }

    // ---- scale + causal mask + online softmax ----
    float p0[8], p1[8], tm[8];
#pragma unroll
    for (int r = 0; r < 8; ++r) {
      const int rowg = qlo + r + off8;
      const float v0 = (j0 + lo      > rowg) ? -__builtin_inff() : s[0][r] * scale;
      const float v1 = (j0 + 16 + lo > rowg) ? -__builtin_inff() : s[1][r] * scale;
      s[0][r] = v0; s[1][r] = v1;
      tm[r] = fmaxf(v0, v1);
    }
#pragma unroll
    for (int m = 1; m < 16; m <<= 1)
#pragma unroll
      for (int r = 0; r < 8; ++r) tm[r] = fmaxf(tm[r], __shfl_xor(tm[r], m, 32));

    float csc[8], ts[8];
#pragma unroll
    for (int r = 0; r < 8; ++r) {
      const float mn = fmaxf(mstate[r], tm[r]);
      csc[r] = __expf(mstate[r] - mn);
      mstate[r] = mn;
      p0[r] = __expf(s[0][r] - mn);
      p1[r] = __expf(s[1][r] - mn);
      ts[r] = p0[r] + p1[r];
    }
#pragma unroll
    for (int m = 1; m < 16; m <<= 1)
#pragma unroll
      for (int r = 0; r < 8; ++r) ts[r] += __shfl_xor(ts[r], m, 32);
#pragma unroll
    for (int r = 0; r < 8; ++r) lstate[r] = lstate[r] * csc[r] + ts[r];
#pragma unroll
    for (int t = 0; t < 4; ++t)
#pragma unroll
      for (int r = 0; r < 8; ++r) o[t][r] = o[t][r] * csc[r];

    // ---- transpose P (C-layout) -> A-fragment via LDS ----
#pragma unroll
    for (int r = 0; r < 8; ++r) {
      P[(r + off8) * 32 + lo]      = f2bf(p0[r]);
      P[(r + off8) * 32 + 16 + lo] = f2bf(p1[r]);
    }
    __syncthreads();
    bf16x16 pa;
    {
      bf16x8 l = *(const bf16x8*)&P[lo * 32 + ah];
      bf16x8 h = *(const bf16x8*)&P[lo * 32 + 8 + ah];
#pragma unroll
      for (int i = 0; i < 8; ++i) { pa[i] = l[i]; pa[8 + i] = h[i]; }
    }

    // ---- O += P @ V  (V pre-transposed: contiguous K per lane) ----
    bf16x16 vb[4];
#pragma unroll
    for (int td = 0; td < 4; ++td)
      vb[td] = *(const bf16x16*)(Vt + ((size_t)bh * HD + td * 16 + lo) * TT +
                                 j0 + bh2);
#pragma unroll
    for (int td = 0; td < 4; ++td)
      o[td] = __builtin_amdgcn_wmma_f32_16x16x32_bf16(
          false, pa, false, vb[td], (short)0, o[td], false, false);
    __syncthreads();
  }

  // ---- normalize and write out to [b, t, h*HD + d] (bf16) ----
  const int b = bh / NH, h = bh % NH;
#pragma unroll
  for (int td = 0; td < 4; ++td)
#pragma unroll
    for (int r = 0; r < 8; ++r) {
      const size_t row = (size_t)b * TT + qlo + r + off8;
      O[row * DIMC + h * HD + td * 16 + lo] = f2bf(o[td][r] / lstate[r]);
    }
}

// ---------------------------------------------------------------------------
extern "C" void kernel_launch(void* const* d_in, const int* in_sizes, int n_in,
                              void* d_out, int out_size, void* d_ws, size_t ws_size,
                              hipStream_t stream) {
  (void)in_sizes; (void)n_in; (void)out_size; (void)ws_size;
  const float* x      = (const float*)d_in[0];
  const float* w_qkv  = (const float*)d_in[1];
  const float* w_proj = (const float*)d_in[2];
  const float* qnw    = (const float*)d_in[3];
  const float* knw    = (const float*)d_in[4];
  float* out = (float*)d_out;

  // workspace layout
  constexpr size_t OFF_QKVF = 0;                                   // f32 [4096,3072]
  constexpr size_t OFF_XB   = OFF_QKVF + (size_t)4096 * 3072 * 4;  // bf16 [4096,1024]
  constexpr size_t OFF_WQ   = OFF_XB   + (size_t)4096 * 1024 * 2;  // bf16 [3072,1024]
  constexpr size_t OFF_WP   = OFF_WQ   + (size_t)3072 * 1024 * 2;  // bf16 [1024,1024]
  constexpr size_t OFF_QB   = OFF_WP   + (size_t)1024 * 1024 * 2;  // bf16 [32,2048,64]
  constexpr size_t OFF_KB   = OFF_QB   + (size_t)32 * 2048 * 64 * 2;
  constexpr size_t OFF_VT   = OFF_KB   + (size_t)32 * 2048 * 64 * 2;
  constexpr size_t OFF_AO   = OFF_VT   + (size_t)32 * 2048 * 64 * 2;

  char* ws = (char*)d_ws;
  float*  qkvf  = (float*) (ws + OFF_QKVF);
  __bf16* xb    = (__bf16*)(ws + OFF_XB);
  __bf16* wqkvb = (__bf16*)(ws + OFF_WQ);
  __bf16* wprjb = (__bf16*)(ws + OFF_WP);
  __bf16* qb    = (__bf16*)(ws + OFF_QB);
  __bf16* kbuf  = (__bf16*)(ws + OFF_KB);
  __bf16* vT    = (__bf16*)(ws + OFF_VT);
  __bf16* aout  = (__bf16*)(ws + OFF_AO);

  // 1) convert inputs to bf16
  {
    int n = BB * TT * DIMC;                                  // 4,194,304
    attn_cvt_bf16<<<n / 256, 256, 0, stream>>>(x, xb, n);
    n = 3 * DIMC * DIMC;                                     // 3,145,728
    attn_cvt_bf16<<<n / 256, 256, 0, stream>>>(w_qkv, wqkvb, n);
    n = DIMC * DIMC;                                         // 1,048,576
    attn_cvt_bf16<<<n / 256, 256, 0, stream>>>(w_proj, wprjb, n);
  }

  // 2) qkv = x @ w_qkv^T
  {
    dim3 g(3 * DIMC / 64, BB * TT / 32);                     // (48, 128)
    attn_gemm_bf16<<<g, 32, 0, stream>>>(xb, wqkvb, qkvf, BB * TT, 3 * DIMC, DIMC);
  }

  // 3) RMSNorm + RoPE -> bf16 q, k, v^T
  attn_qkv_post<<<BB * NH * TT, 32, 0, stream>>>(qkvf, qnw, knw, qb, kbuf, vT);

  // 4) causal flash attention
  {
    dim3 g(BB * NH, TT / 16);                                // (32, 128)
    attn_flash<<<g, 32, 0, stream>>>(qb, kbuf, vT, aout);
  }

  // 5) out = attn_out @ w_proj^T  (f32 result)
  {
    dim3 g(DIMC / 64, BB * TT / 32);                         // (16, 128)
    attn_gemm_bf16<<<g, 32, 0, stream>>>(aout, wprjb, out, BB * TT, DIMC, DIMC);
  }
}
